// Attn_39247411151213
// MI455X (gfx1250) — compile-verified, matrix operand order checked by reading, not run on hardware
//
#include <hip/hip_runtime.h>
#include <hip/hip_bf16.h>

// ---------------------------------------------------------------------------
// Fused MHA (heads-last interleave) for MI455X / gfx1250, wave32 + WMMA f16.
// Stage 1: Q/K/V projections via v_wmma_f32_16x16x32_f16, de-interleaved into
//          [B,H,S,dh] f16 workspace (bias fused for Q).
// Stage 2: flash attention; Q/K tiles streamed with GLOBAL_LOAD_ASYNC_TO_LDS
//          (ASYNCcnt-tracked, K double-buffered), both matmuls on WMMA,
//          fp32 online softmax, P staged through LDS for the layout swap.
// ---------------------------------------------------------------------------

typedef __attribute__((ext_vector_type(16))) _Float16 v16h;
typedef __attribute__((ext_vector_type(8)))  _Float16 v8h;
typedef __attribute__((ext_vector_type(8)))  float    v8f;

#define WMMA_F16(a, b, c) \
    __builtin_amdgcn_wmma_f32_16x16x32_f16(false, (a), false, (b), (short)0, (c), false, false)

// Per-lane 16B global->LDS async copy (CDNA5 VGLOBAL async op, ASYNCcnt).
static __device__ __forceinline__
void async_copy_b128(unsigned lds_addr, const _Float16* g) {
    asm volatile("global_load_async_to_lds_b128 %0, %1, off"
                 :: "v"(lds_addr), "v"((unsigned long long)(uintptr_t)g)
                 : "memory");
}
#define WAIT_ASYNCCNT(n) asm volatile("s_wait_asynccnt " #n ::: "memory")

// Low 32 bits of a generic pointer to LDS == LDS byte offset (ISA §10.2).
static __device__ __forceinline__ unsigned lds_off(const void* p) {
    return (unsigned)(uintptr_t)p;
}

// Load a 16x32 f16 A/B fragment from an LDS tile stored [row][k] (rstride
// halves per row). ISA 16-bit A layout: lanes 0-15 hold K {0..7,16..23},
// lanes 16-31 hold K {8..15,24..31} (wave32).
static __device__ __forceinline__
v16h ldfrag(const _Float16* __restrict__ base, int row, int rstride, int koff, int lane) {
    const _Float16* p = base + row * rstride + koff + ((lane & 16) ? 8 : 0);
    v8h lo = *(const v8h*)(p);
    v8h hi = *(const v8h*)(p + 16);
    v16h r;
#pragma unroll
    for (int i = 0; i < 8; ++i) { r[i] = lo[i]; r[i + 8] = hi[i]; }
    return r;
}

// ---------------------------------------------------------------------------
// Stage 1: QKV projection GEMM.  out[z] = x @ W[z] (+ Bq for z==0)
// x:[4096,1024] f32, W:[1024,1024] f32 -> f16 [B,H,S,dh] (heads de-interleaved)
// Block: 128 threads (4 waves, 2x2), 64x64 output tile, K-step 32.
// (f32->f16 conversion must pass through VGPRs, so tiles load synchronously.)
// ---------------------------------------------------------------------------
__global__ __launch_bounds__(128)
void qkv_proj_kernel(const float* __restrict__ x,
                     const float* __restrict__ Wq,
                     const float* __restrict__ Bq,
                     const float* __restrict__ Wk,
                     const float* __restrict__ Wv,
                     _Float16* __restrict__ Qh,
                     _Float16* __restrict__ Kh,
                     _Float16* __restrict__ Vh) {
    constexpr int D = 1024, S = 2048, H = 16, DH = 64;
    __shared__ __align__(16) _Float16 Al[64 * 32];   // x tile  [m][k]
    __shared__ __align__(16) _Float16 Bt[64 * 32];   // W tile transposed [n][k]

    const int tid   = threadIdx.x;
    const int lane  = tid & 31;
    const int wave  = tid >> 5;
    const int wm    = (wave >> 1) * 32;
    const int wn    = (wave & 1) * 32;
    const int mbase = blockIdx.y * 64;
    const int nbase = blockIdx.x * 64;
    const int z     = blockIdx.z;

    const float* W   = (z == 0) ? Wq : (z == 1) ? Wk : Wv;
    _Float16*    Out = (z == 0) ? Qh : (z == 1) ? Kh : Vh;

    const v8f zv = {0.f, 0.f, 0.f, 0.f, 0.f, 0.f, 0.f, 0.f};
    v8f acc[2][2];
#pragma unroll
    for (int i = 0; i < 2; ++i)
#pragma unroll
        for (int j = 0; j < 2; ++j) acc[i][j] = zv;

    const int lane15 = lane & 15;

    for (int k0 = 0; k0 < D; k0 += 32) {
        // x tile: 64 rows x 32 cols = 512 float4 chunks, 4 per thread
#pragma unroll
        for (int it = 0; it < 4; ++it) {
            int idx = tid + it * 128;
            int row = idx >> 3, c4 = (idx & 7) * 4;
            const float4 v = *(const float4*)(x + (size_t)(mbase + row) * D + k0 + c4);
            _Float16* d = Al + row * 32 + c4;
            d[0] = (_Float16)v.x; d[1] = (_Float16)v.y;
            d[2] = (_Float16)v.z; d[3] = (_Float16)v.w;
        }
        // W tile: 32 rows(k) x 64 cols(n), store transposed [n][k]
#pragma unroll
        for (int it = 0; it < 4; ++it) {
            int idx = tid + it * 128;
            int row = idx >> 4, c4 = (idx & 15) * 4;
            const float4 v = *(const float4*)(W + (size_t)(k0 + row) * D + nbase + c4);
            Bt[(c4 + 0) * 32 + row] = (_Float16)v.x;
            Bt[(c4 + 1) * 32 + row] = (_Float16)v.y;
            Bt[(c4 + 2) * 32 + row] = (_Float16)v.z;
            Bt[(c4 + 3) * 32 + row] = (_Float16)v.w;
        }
        if (k0 + 32 < D)  // speculative prefetch of next x tile row
            __builtin_prefetch(x + (size_t)(mbase + (tid >> 1)) * D + k0 + 32, 0, 1);
        __syncthreads();

        v16h a0 = ldfrag(Al, wm + 0  + lane15, 32, 0, lane);
        v16h a1 = ldfrag(Al, wm + 16 + lane15, 32, 0, lane);
        v16h b0 = ldfrag(Bt, wn + 0  + lane15, 32, 0, lane);
        v16h b1 = ldfrag(Bt, wn + 16 + lane15, 32, 0, lane);
        acc[0][0] = WMMA_F16(a0, b0, acc[0][0]);
        acc[0][1] = WMMA_F16(a0, b1, acc[0][1]);
        acc[1][0] = WMMA_F16(a1, b0, acc[1][0]);
        acc[1][1] = WMMA_F16(a1, b1, acc[1][1]);
        __syncthreads();
    }

    // Epilogue: +bias (Q only), de-interleave heads: col n -> (h = n%16, dq = n/16)
    const int half8 = (lane & 16) ? 8 : 0;
#pragma unroll
    for (int tm = 0; tm < 2; ++tm)
#pragma unroll
        for (int tn = 0; tn < 2; ++tn) {
            const int n    = nbase + wn + tn * 16 + lane15;
            const float bi = (z == 0) ? Bq[n] : 0.0f;
            const int h = n & 15, dq = n >> 4;
#pragma unroll
            for (int r = 0; r < 8; ++r) {
                const int m  = mbase + wm + tm * 16 + r + half8;
                const int bb = m >> 11;          // batch
                const int s  = m & 2047;         // seq pos
                Out[((size_t)(bb * H + h) * S + s) * DH + dq] =
                    (_Float16)(acc[tm][tn][r] + bi);
            }
        }
}

// ---------------------------------------------------------------------------
// Stage 2: flash attention, one block per (64 q-rows, head, batch).
// 128 threads = 4 waves; wave w owns query rows [16w, 16w+16).
// Q + double-buffered K tiles arrive via async global->LDS copies.
// ---------------------------------------------------------------------------
__global__ __launch_bounds__(128)
void flash_attn_kernel(const _Float16* __restrict__ Qh,
                       const _Float16* __restrict__ Kh,
                       const _Float16* __restrict__ Vh,
                       float* __restrict__ out) {
    constexpr int S = 2048, H = 16, DH = 64, D = 1024;
    constexpr int NT = S / 64;          // 32 K/V tiles
    constexpr float SCALE = 0.03125f;   // 1/sqrt(D) = 1/32

    __shared__ __align__(16) _Float16 Qs[64 * 64];       // [qrow][dh]
    __shared__ __align__(16) _Float16 Kb[2][64 * 64];    // [j][dh] double buffer
    __shared__ __align__(16) _Float16 Vt[64 * 64];       // [dh][j] (B of PV)
    __shared__ __align__(16) _Float16 Ps[4][16 * 64];    // per-wave P [row][j]

    const int tid = threadIdx.x, lane = tid & 31, wave = tid >> 5;
    const int lane15 = lane & 15;
    const int half8  = (lane & 16) ? 8 : 0;
    const int q0 = blockIdx.x * 64;
    const int h  = blockIdx.y;
    const int b  = blockIdx.z;

    const size_t headoff = (size_t)(b * H + h) * S * DH;
    const _Float16* Qg = Qh + headoff;
    const _Float16* Kg = Kh + headoff;
    const _Float16* Vg = Vh + headoff;

    const unsigned ldsQ  = lds_off(Qs);
    const unsigned ldsK0 = lds_off(Kb[0]);
    const unsigned ldsK1 = lds_off(Kb[1]);

    // Kick off async copies: K tile 0 first, then the resident Q tile.
    // Each tile is 8KB contiguous: 512 x 16B chunks, 4 per thread.
#pragma unroll
    for (int it = 0; it < 4; ++it) {
        int idx = tid + it * 128;
        async_copy_b128(ldsK0 + idx * 16, Kg + idx * 8);
    }
#pragma unroll
    for (int it = 0; it < 4; ++it) {
        int idx = tid + it * 128;
        async_copy_b128(ldsQ + idx * 16, Qg + (size_t)q0 * DH + idx * 8);
    }
    WAIT_ASYNCCNT(0x0);   // Q and K0 resident in this wave's view

    const v8f zv = {0.f, 0.f, 0.f, 0.f, 0.f, 0.f, 0.f, 0.f};
    v8f o[4];
#pragma unroll
    for (int t = 0; t < 4; ++t) o[t] = zv;
    float m_st[8], l_st[8];
#pragma unroll
    for (int r = 0; r < 8; ++r) { m_st[r] = -1e30f; l_st[r] = 0.0f; }

    for (int ti = 0; ti < NT; ++ti) {
        const int j0 = ti * 64;
        __syncthreads();  // previous iteration done reading Vt / K buffer

        // V tile transposed -> Vt[dh][j] (needs VGPR pass, regular loads)
#pragma unroll
        for (int it = 0; it < 4; ++it) {
            int idx = tid + it * 128;
            int row = idx >> 3, c = (idx & 7) * 8;
            v8h v = *(const v8h*)(Vg + (size_t)(j0 + row) * DH + c);
#pragma unroll
            for (int u = 0; u < 8; ++u) Vt[(c + u) * 64 + row] = v[u];
        }

        // Prefetch K tile ti+1 into the other buffer (read last in iter ti-1,
        // so the barrier above protects it), then wait for K tile ti.
        if (ti + 1 < NT) {
            const unsigned dst = ((ti + 1) & 1) ? ldsK1 : ldsK0;
#pragma unroll
            for (int it = 0; it < 4; ++it) {
                int idx = tid + it * 128;
                async_copy_b128(dst + idx * 16, Kg + (size_t)(j0 + 64) * DH + idx * 8);
            }
            WAIT_ASYNCCNT(0x4);   // in-order: first 4 outstanding (tile ti) done
        } else {
            WAIT_ASYNCCNT(0x0);   // last tile: drain
        }
        __syncthreads();          // Vt stores + K tile ti visible block-wide

        const _Float16* Kcur = Kb[ti & 1];

        // ---- S = Q K^T  (contraction over dh; K[j][dh] is exactly B[k][n]) ----
        v8f sc[4];
#pragma unroll
        for (int t = 0; t < 4; ++t) sc[t] = zv;
#pragma unroll
        for (int kk = 0; kk < 64; kk += 32) {
            v16h a = ldfrag(Qs, wave * 16 + lane15, 64, kk, lane);
#pragma unroll
            for (int tn = 0; tn < 4; ++tn) {
                v16h bk = ldfrag(Kcur, tn * 16 + lane15, 64, kk, lane);
                sc[tn] = WMMA_F16(a, bk, sc[tn]);
            }
        }
#pragma unroll
        for (int tn = 0; tn < 4; ++tn)
#pragma unroll
            for (int r = 0; r < 8; ++r) sc[tn][r] *= SCALE;

        // ---- online softmax (row m = r + half8 lives across 16 lanes) ----
        float corr[8];
#pragma unroll
        for (int r = 0; r < 8; ++r) {
            float mx = fmaxf(fmaxf(sc[0][r], sc[1][r]), fmaxf(sc[2][r], sc[3][r]));
#pragma unroll
            for (int off = 1; off < 16; off <<= 1)
                mx = fmaxf(mx, __shfl_xor(mx, off, 32));
            float nm = fmaxf(m_st[r], mx);
            corr[r]  = __expf(m_st[r] - nm);
            m_st[r]  = nm;
        }
        float rs[8];
#pragma unroll
        for (int r = 0; r < 8; ++r) rs[r] = 0.0f;
#pragma unroll
        for (int tn = 0; tn < 4; ++tn)
#pragma unroll
            for (int r = 0; r < 8; ++r) {
                float p = __expf(sc[tn][r] - m_st[r]);
                sc[tn][r] = p;
                rs[r] += p;
            }
#pragma unroll
        for (int r = 0; r < 8; ++r) {
            float t = rs[r];
#pragma unroll
            for (int off = 1; off < 16; off <<= 1) t += __shfl_xor(t, off, 32);
            l_st[r] = l_st[r] * corr[r] + t;
#pragma unroll
            for (int tn = 0; tn < 4; ++tn) o[tn][r] *= corr[r];
        }

        // ---- P: C-layout -> A-layout via per-wave LDS (in-order DS, same wave) ----
        _Float16* P = Ps[wave];
#pragma unroll
        for (int tn = 0; tn < 4; ++tn)
#pragma unroll
            for (int r = 0; r < 8; ++r)
                P[(r + half8) * 64 + tn * 16 + lane15] = (_Float16)sc[tn][r];
        __builtin_amdgcn_wave_barrier();   // keep DS stores before DS frag loads

        // ---- O += P V  (contraction over j; Vt[dh][j] is B[k][n]) ----
#pragma unroll
        for (int kk = 0; kk < 64; kk += 32) {
            v16h a = ldfrag(P, lane15, 64, kk, lane);
#pragma unroll
            for (int tn = 0; tn < 4; ++tn) {
                v16h bv = ldfrag(Vt, tn * 16 + lane15, 64, kk, lane);
                o[tn] = WMMA_F16(a, bv, o[tn]);
            }
        }
    }

    // ---- epilogue: O / l, re-interleave heads (out col = dq*16 + h) ----
#pragma unroll
    for (int tn = 0; tn < 4; ++tn) {
        const int dq = tn * 16 + lane15;
#pragma unroll
        for (int r = 0; r < 8; ++r) {
            const int srow = q0 + wave * 16 + r + half8;
            out[((size_t)b * S + srow) * D + dq * H + h] = o[tn][r] / l_st[r];
        }
    }
}

// ---------------------------------------------------------------------------
extern "C" void kernel_launch(void* const* d_in, const int* in_sizes, int n_in,
                              void* d_out, int out_size, void* d_ws, size_t ws_size,
                              hipStream_t stream) {
    const float* x  = (const float*)d_in[0];
    const float* Wq = (const float*)d_in[1];
    const float* Bq = (const float*)d_in[2];
    const float* Wk = (const float*)d_in[3];
    const float* Wv = (const float*)d_in[4];
    float* out = (float*)d_out;

    // f16 workspace: Q/K/V in [B,H,S,dh] = 4,194,304 halves (8 MB) each
    const size_t per = (size_t)2 * 16 * 2048 * 64;
    _Float16* Qh = (_Float16*)d_ws;
    _Float16* Kh = Qh + per;
    _Float16* Vh = Kh + per;

    dim3 g1(1024 / 64, 4096 / 64, 3);   // (N tiles, M tiles, {Q,K,V})
    qkv_proj_kernel<<<g1, 128, 0, stream>>>(x, Wq, Bq, Wk, Wv, Qh, Kh, Vh);

    dim3 g2(2048 / 64, 16, 2);          // (q tiles, heads, batch)
    flash_attn_kernel<<<g2, 128, 0, stream>>>(Qh, Kh, Vh, out);
}
